// PersistentGraphConv_56573309223828
// MI455X (gfx1250) — matrix operand order repaired and checked by reading, not compiled, not run on hardware
//
#include <hip/hip_runtime.h>
#include <hip/hip_bf16.h>
#include <stdint.h>

typedef __attribute__((ext_vector_type(16))) _Float16 v16h;
typedef __attribute__((ext_vector_type(8)))  _Float16 v8h;
typedef __attribute__((ext_vector_type(8)))  float    v8f;
typedef __attribute__((ext_vector_type(4)))  unsigned int ui32x4;
typedef __attribute__((ext_vector_type(8)))  int      i32x8;
typedef __attribute__((ext_vector_type(4)))  int      i32x4;

#define IN_CH  128
#define OUT_CH 64

// LDS pitches:
//  Xs (f32): pitch 132 floats -> lane bank-stride 4; each 16-lane phase of a
//            b128 A-operand load covers all 64 banks exactly once.
//  Wt (f16): pitch 160 halves -> lane bank-stride 16; the two 32B halves of the
//            B operand land on disjoint bank octets (conflict-free).
#define XS_PITCH 132
#define WT_PITCH 160

// ---------------------------------------------------------------------------
// Kernel 1: out[i] = b[i%64] (bias pre-init, harness poisons d_out), deg = 0
// ---------------------------------------------------------------------------
__global__ __launch_bounds__(256) void gcn_init_kernel(
    float* __restrict__ out, float* __restrict__ deg,
    const float* __restrict__ b, long long total_out, int n_nodes) {
  long long i = (long long)blockIdx.x * 256 + threadIdx.x;
  if (i < total_out) out[i] = b[(int)(i & 63)];
  if (i < n_nodes)   deg[i] = 0.0f;
}

// ---------------------------------------------------------------------------
// Kernel 2: h = x @ W via v_wmma_f32_16x16x32_f16.
// Block = 256 threads = 8 waves; block computes 32 rows x 64 cols.
// x tile (32x128 f32, 16.5KB padded) is DMA'd into LDS by the Tensor Data
// Mover (one tensor_load_to_lds per block, TENSORcnt-tracked); TDM padding
// inserts 4 dwords per 128-dword row to realize the 132-float LDS pitch.
// ---------------------------------------------------------------------------
__global__ __launch_bounds__(256) void gcn_gemm_kernel(
    const float* __restrict__ x, const float* __restrict__ W,
    float* __restrict__ h, int n_nodes) {
  __shared__ float    Xs[32 * XS_PITCH];        // 32 x-rows, f32, pitch 132
  __shared__ _Float16 WtH[OUT_CH * WT_PITCH];   // Wt[n][k] = W[k][n], f16

  const int t = threadIdx.x;
  const int block_row0 = blockIdx.x * 32;

  // ---- TDM path: one descriptor DMAs the whole 32x128 f32 tile ----
  if (t < 32) {  // wave 0 issues (EXEC is ignored by tensor ops; one wave = one issue)
    const unsigned long long gaddr =
        (unsigned long long)(uintptr_t)(x + (size_t)block_row0 * IN_CH);
    const unsigned int lds_off = (unsigned int)(uintptr_t)&Xs[0];
    const unsigned int tdim1   = (unsigned int)(n_nodes - block_row0); // rows left (OOB reads -> 0)

    ui32x4 g0;
    g0[0] = 1u;                                              // count=1 (valid), user mode
    g0[1] = lds_off;                                         // lds_addr [63:32]
    g0[2] = (unsigned int)(gaddr & 0xFFFFFFFFu);             // global_addr [95:64]
    g0[3] = (unsigned int)((gaddr >> 32) & 0x01FFFFFFu)      // global_addr [120:96]
          | (2u << 30);                                      // type=2 ("image") [127:126]

    i32x8 g1;
    g1[0] = (int)((2u << 16)      // data_size = 4 bytes
                | (1u << 20)      // pad_enable
                | (6u << 22)      // pad_interval: 128 dwords (row length)
                | (3u << 25));    // pad_amount: 4 dwords -> pitch 132 floats
    g1[1] = (int)(128u << 16);    // tensor_dim0 = 128      [79:48] low half
    g1[2] = (int)((tdim1 & 0xFFFFu) << 16);                 // tensor_dim1 [95:80]
    g1[3] = (int)((tdim1 >> 16) | (128u << 16));            // tensor_dim1 hi, tile_dim0=128
    g1[4] = (int)32u;             // tile_dim1 = 32, tile_dim2 = 0
    g1[5] = (int)128u;            // tensor_dim0_stride = 128 (low 32)
    g1[6] = 0;                    // stride hi, tensor_dim1_stride (unused, 2D)
    g1[7] = 0;

    const i32x4 gz4 = {0, 0, 0, 0};                  // groups 2/3 unused (2D tensor)
    const i32x8 gz8 = {0, 0, 0, 0, 0, 0, 0, 0};      // extra group (clang-23 6-arg form)
    __builtin_amdgcn_tensor_load_to_lds(g0, g1, gz4, gz4, gz8, 0);
  }

  // Stage W transposed into LDS as f16 (coalesced global read of W[k][n]);
  // overlaps with the TDM transfer.
  for (int idx = t; idx < IN_CH * OUT_CH; idx += 256) {
    const int k = idx >> 6;    // 0..127
    const int n = idx & 63;    // 0..63
    WtH[n * WT_PITCH + k] = (_Float16)W[idx];
  }

  if (t < 32) __builtin_amdgcn_s_wait_tensorcnt(0);  // issuing wave drains TDM
  __syncthreads();

  const int wid  = t >> 5;
  const int lane = t & 31;
  const bool hi  = lane >= 16;
  const int m    = lane & 15;        // row-within-tile (A) / col-within-tile (B,D)
  const int rt   = wid >> 2;         // 0..1
  const int ct   = wid & 3;          // 0..3
  const int n0   = ct * 16;

  const float*    aRow = &Xs[(rt * 16 + m) * XS_PITCH];
  const _Float16* bCol = &WtH[(n0 + m) * WT_PITCH];
  const int kbA = hi ? 16 : 0;       // A: lanes 16-31 hold K = 16..31 of the step
  const int kbB = hi ? 16 : 0;       // B: lanes 16-31 hold K = 16..31 of the step

  v8f acc = {};
#pragma unroll
  for (int ks = 0; ks < 4; ++ks) {
    const int k0 = ks * 32;
    // A operand: element j -> K = k0 + kbA + j
    const float4 f0 = *(const float4*)(aRow + k0 + kbA);
    const float4 f1 = *(const float4*)(aRow + k0 + kbA + 4);
    const float4 f2 = *(const float4*)(aRow + k0 + kbA + 8);
    const float4 f3 = *(const float4*)(aRow + k0 + kbA + 12);
    v16h a;
    a[0]  = (_Float16)f0.x; a[1]  = (_Float16)f0.y;
    a[2]  = (_Float16)f0.z; a[3]  = (_Float16)f0.w;
    a[4]  = (_Float16)f1.x; a[5]  = (_Float16)f1.y;
    a[6]  = (_Float16)f1.z; a[7]  = (_Float16)f1.w;
    a[8]  = (_Float16)f2.x; a[9]  = (_Float16)f2.y;
    a[10] = (_Float16)f2.z; a[11] = (_Float16)f2.w;
    a[12] = (_Float16)f3.x; a[13] = (_Float16)f3.y;
    a[14] = (_Float16)f3.z; a[15] = (_Float16)f3.w;
    // B operand: element j -> K = k0 + kbB + j, column n0+m
    const v16h bm = *(const v16h*)(bCol + k0 + kbB);
    acc = __builtin_amdgcn_wmma_f32_16x16x32_f16(
        /*neg_a=*/false, a, /*neg_b=*/false, bm,
        /*c_mod=*/(short)0, acc, /*reuse_a=*/false, /*reuse_b=*/false);
  }

  // D layout: VGPR r holds M = r (+8 for lanes 16-31), N = lane&15.
  const int n     = n0 + m;
  const int mrow0 = block_row0 + rt * 16 + (hi ? 8 : 0);
  float* hp = h + (size_t)mrow0 * OUT_CH + n;
  if (block_row0 + 32 <= n_nodes) {      // wave-uniform fast path: no guards
#pragma unroll
    for (int r = 0; r < 8; ++r) hp[(size_t)r * OUT_CH] = acc[r];
  } else {                               // tail block (unused when N % 32 == 0)
#pragma unroll
    for (int r = 0; r < 8; ++r)
      if (mrow0 + r < n_nodes) hp[(size_t)r * OUT_CH] = acc[r];
  }
}

// ---------------------------------------------------------------------------
// Kernel 3: deg[row[e]] += edge_weight[e]  (HW fp32 atomics)
// ---------------------------------------------------------------------------
__global__ __launch_bounds__(256) void gcn_deg_kernel(
    const long long* __restrict__ edge_index, const float* __restrict__ ew,
    float* __restrict__ deg, int n_edges) {
  const int e = blockIdx.x * 256 + threadIdx.x;
  if (e < n_edges) {
    const int r = (int)edge_index[e];           // row = edge_index[0][e]
    unsafeAtomicAdd(&deg[r], ew[e]);
  }
}

// ---------------------------------------------------------------------------
// Kernel 4: deg -> deg^-0.5 in place (0 where deg == 0)
// ---------------------------------------------------------------------------
__global__ __launch_bounds__(256) void gcn_rsqrt_kernel(
    float* __restrict__ deg, int n_nodes) {
  const int i = blockIdx.x * 256 + threadIdx.x;
  if (i < n_nodes) {
    const float d = deg[i];
    deg[i] = (d == 0.0f) ? 0.0f : rsqrtf(d);
  }
}

// ---------------------------------------------------------------------------
// Kernel 5: out[row] += h[col] * dis[row]*dis[col]   (64 threads per edge,
// coalesced 256B h-row gathers, HW fp32 atomics into out)
// ---------------------------------------------------------------------------
__global__ __launch_bounds__(256) void gcn_scatter_kernel(
    const long long* __restrict__ edge_index, const float* __restrict__ h,
    const float* __restrict__ dis, float* __restrict__ out, int n_edges) {
  const long long tid = (long long)blockIdx.x * 256 + threadIdx.x;
  const int e = (int)(tid >> 6);
  const int c = (int)(tid & 63);
  if (e < n_edges) {
    const int r   = (int)edge_index[e];                     // destination
    const int col = (int)edge_index[(size_t)n_edges + e];   // source
    const float norm = dis[r] * dis[col];
    const float val  = h[(size_t)col * OUT_CH + c] * norm;
    unsafeAtomicAdd(&out[(size_t)r * OUT_CH + c], val);
  }
}

// ---------------------------------------------------------------------------
extern "C" void kernel_launch(void* const* d_in, const int* in_sizes, int n_in,
                              void* d_out, int out_size, void* d_ws, size_t ws_size,
                              hipStream_t stream) {
  const float*     x  = (const float*)d_in[0];       // [N, 128] f32
  const long long* ei = (const long long*)d_in[1];   // [2, E]  i64
  const float*     ew = (const float*)d_in[2];       // [E]     f32
  const float*     W  = (const float*)d_in[3];       // [128,64] f32
  const float*     b  = (const float*)d_in[4];       // [64]    f32
  float* out = (float*)d_out;                        // [N, 64] f32

  const int n_nodes = in_sizes[0] / IN_CH;
  const int n_edges = in_sizes[2];

  // Workspace: h [N*64] f32, then deg [N] f32.
  float* h   = (float*)d_ws;
  float* deg = h + (size_t)n_nodes * OUT_CH;

  const long long total_out = (long long)n_nodes * OUT_CH;
  gcn_init_kernel<<<(int)((total_out + 255) / 256), 256, 0, stream>>>(
      out, deg, b, total_out, n_nodes);

  gcn_gemm_kernel<<<(n_nodes + 31) / 32, 256, 0, stream>>>(x, W, h, n_nodes);

  gcn_deg_kernel<<<(n_edges + 255) / 256, 256, 0, stream>>>(ei, ew, deg, n_edges);

  gcn_rsqrt_kernel<<<(n_nodes + 255) / 256, 256, 0, stream>>>(deg, n_nodes);

  const long long total_e = (long long)n_edges * OUT_CH;
  gcn_scatter_kernel<<<(int)((total_e + 255) / 256), 256, 0, stream>>>(
      ei, h, deg, out, n_edges);
}
